// myconv1_4939212390738
// MI455X (gfx1250) — matrix operand order repaired and checked by reading, not compiled, not run on hardware
//
#include <hip/hip_runtime.h>
#include <hip/hip_bf16.h>
#include <stdint.h>

// ---------------------------------------------------------------------------
// MI455X (gfx1250) implementation, round 2.
//  - Both 3x3 convs as implicit-GEMM on V_WMMA_F32_16X16X32_BF16.
//  - Activations pre-padded NHWC bf16 -> A-tile fill is 1 b128 load +
//    1 ds_store_b128 per thread, no bounds checks in the hot loop.
//  - Weights pre-tiled into the exact LDS image -> B-tile fill likewise.
//  - LDS ping-pong double buffering: one barrier per K-step, global loads
//    for step s+1 overlap WMMAs of step s.
//  - f = conv3(x)*sigmoid(x+up(y1)) folded into the conv epilogue, reduced
//    straight to gm (B x C): the 64 MiB f tensor is never materialized.
// ---------------------------------------------------------------------------

typedef __attribute__((ext_vector_type(16))) __bf16 v16bf;
typedef __attribute__((ext_vector_type(8)))  float  v8f;

__device__ __forceinline__ unsigned short f2bf(float f) {
  uint32_t u = __float_as_uint(f);
  uint32_t r = (u + 0x7FFFu + ((u >> 16) & 1u)) >> 16;  // RNE bf16
  return (unsigned short)r;
}

// ------------------------- prep kernels ------------------------------------

__global__ __launch_bounds__(256) void mk_zero_u32(uint32_t* __restrict__ p) {
  p[(size_t)blockIdx.x * 256 + threadIdx.x] = 0u;
}

// x (B,C,64,64) fp32 -> xpad (B,66,66,C) bf16 interior; borders pre-zeroed.
// block = (b*64 + h), 256 threads = c; writes are 512B-contiguous.
__global__ __launch_bounds__(256) void mk_x_nhwc(const float* __restrict__ x,
                                                 unsigned short* __restrict__ xpad) {
  const int c = threadIdx.x;
  const int h = blockIdx.x & 63;
  const int b = blockIdx.x >> 6;
  const float* src = x + (((size_t)b * 256 + c) * 64 + h) * 64;
  unsigned short* dst = xpad + (((size_t)b * 66 + h + 1) * 66 + 1) * 256 + c;
  for (int w = 0; w < 64; ++w) dst[w * 256] = f2bf(src[w]);
}

// 4x4 avg pool -> y0pad (B,18,18,C) bf16 interior; borders pre-zeroed.
__global__ __launch_bounds__(256) void mk_pool_nhwc(const float* __restrict__ x,
                                                    unsigned short* __restrict__ y0pad) {
  const int c = threadIdx.x;
  const int ph = blockIdx.x & 15;
  const int b = blockIdx.x >> 4;
  const float* src = x + (((size_t)b * 256 + c) * 64 + ph * 4) * 64;
  unsigned short* dst = y0pad + (((size_t)b * 18 + ph + 1) * 18 + 1) * 256 + c;
  for (int pw = 0; pw < 16; ++pw) {
    float s = 0.f;
#pragma unroll
    for (int r = 0; r < 4; ++r)
#pragma unroll
      for (int q = 0; q < 4; ++q) s += src[r * 64 + pw * 4 + q];
    dst[pw * 256] = f2bf(s * 0.0625f);
  }
}

// (co,ci,3,3) fp32 -> tiled bf16: [(tap*8+ks)*4+co_t][n<64][k<32]
// so the conv B-fill is a contiguous b128 load at base + tid*8.
__global__ __launch_bounds__(256) void mk_wxform(const float* __restrict__ src,
                                                 unsigned short* __restrict__ dst) {
  int idx = blockIdx.x * 256 + threadIdx.x;   // < 9*8*4*64*32 = 589824
  int k    = idx & 31;
  int n    = (idx >> 5) & 63;
  int co_t = (idx >> 11) & 3;
  int ks   = (idx >> 13) & 7;
  int tap  = idx >> 16;
  int co = co_t * 64 + n;
  int ci = ks * 32 + k;
  dst[idx] = f2bf(src[co * 2304 + ci * 9 + tap]);
}

__global__ __launch_bounds__(256) void mk_zero_f32(float* __restrict__ p, int n) {
  int i = blockIdx.x * 256 + threadIdx.x;
  if (i < n) p[i] = 0.f;
}

// ------------------------- WMMA implicit-GEMM conv3x3 ----------------------
// 256 threads = 8 waves (4 M-subtiles x 2 N-subtiles).
// Output tile: 64 spatial positions x 64 output channels.
// 72 K-steps of K=32 (9 taps x 8 channel-slices), LDS ping-pong.
// LDS rows padded to 40 ushort (80B) so b128 stores stay 16B-aligned.
template <int HH, int W, int MODE>
__global__ __launch_bounds__(256) void mk_conv3x3_wmma(
    const unsigned short* __restrict__ Apad,  // bf16 NHWC padded (B,HH+2,W+2,256)
    const unsigned short* __restrict__ WtT,   // bf16 tiled weights
    const float* __restrict__ bias,
    const float* __restrict__ bnw, const float* __restrict__ bnb,
    const float* __restrict__ bnm, const float* __restrict__ bnv,
    const float* __restrict__ xf32,           // MODE1: x fp32 NCHW
    const float* __restrict__ y1in,           // MODE1: y1 fp32 NCHW (B,256,16,16)
    float* __restrict__ outp)                 // MODE0: y1 ; MODE1: gm
{
  constexpr int C  = 256;
  constexpr int HP = HH + 2;
  constexpr int WP = W + 2;
  constexpr int LS = 40;       // LDS row stride in ushort (80 B)
  constexpr int NSTEP = 72;

  __shared__ __align__(16) unsigned short AsBuf[2][64 * LS];
  __shared__ __align__(16) unsigned short BsBuf[2][64 * LS];

  const int tid  = threadIdx.x;
  const int lane = tid & 31;
  const int wave = tid >> 5;
  const int wm   = wave & 3;
  const int wn   = wave >> 2;
  const int co_t  = blockIdx.x;          // 0..3
  const int mbase = blockIdx.y * 64;
  const int b     = blockIdx.z;

  // per-thread fill coordinates (step-invariant parts)
  const int mfill = tid >> 2;            // 0..63 (spatial position in tile)
  const int kseg  = (tid & 3) * 8;       // 8-element channel segment
  const int posm  = mbase + mfill;
  const int hf    = posm / W;
  const int wf    = posm % W;
  const int ldsoff = mfill * LS + kseg;  // same offset for As and Bs fills

  auto ldA = [&](int step) -> uint4 {
    int tap = step >> 3, ks = step & 7;
    size_t off = (((size_t)b * HP + hf + tap / 3) * WP + (wf + tap % 3)) * C
               + ks * 32 + kseg;
    return *reinterpret_cast<const uint4*>(Apad + off);
  };
  auto ldB = [&](int step) -> uint4 {
    size_t off = (((size_t)step * 4 + co_t) << 11) + tid * 8;
    return *reinterpret_cast<const uint4*>(WtT + off);
  };

  v8f acc0 = {};
  v8f acc1 = {};

  // prologue: stage step 0
  {
    uint4 ra = ldA(0), rb = ldB(0);
    *reinterpret_cast<uint4*>(&AsBuf[0][ldsoff]) = ra;
    *reinterpret_cast<uint4*>(&BsBuf[0][ldsoff]) = rb;
  }
  __syncthreads();

  int p = 0;
  for (int step = 0; step < NSTEP; ++step) {
    // prefetch + register-stage next step while WMMAing this one
    uint4 ra, rb;
    const bool more = (step + 1) < NSTEP;
    if (lane == 0 && step + 2 < NSTEP)
      __builtin_prefetch(WtT + (((size_t)(step + 2) * 4 + co_t) << 11), 0, 0);
    if (more) { ra = ldA(step + 1); rb = ldB(step + 1); }

    const uint32_t* As32 = (const uint32_t*)AsBuf[p];
    const uint32_t* Bs32 = (const uint32_t*)BsBuf[p];
    union { uint32_t u[8]; v16bf v; } af, bfr0, bfr1;
    {
      int r  = 16 * wm + (lane & 15);
      int kb = (lane >> 4) * 4;                  // K base (u32): 0 or 4
#pragma unroll
      for (int i = 0; i < 4; ++i) {
        af.u[i]     = As32[r * 20 + kb + i];     // K = 8*half + 2i
        af.u[4 + i] = As32[r * 20 + 8 + kb + i]; // K = 16 + 8*half + 2i
      }
    }
    {
      int n0  = 32 * wn + (lane & 15);
      int kb2 = (lane >> 4) * 4;                 // K base (u32): 0 or 4
#pragma unroll
      for (int i = 0; i < 4; ++i) {
        bfr0.u[i]     = Bs32[n0 * 20 + kb2 + i];
        bfr0.u[4 + i] = Bs32[n0 * 20 + 8 + kb2 + i];
        bfr1.u[i]     = Bs32[(n0 + 16) * 20 + kb2 + i];
        bfr1.u[4 + i] = Bs32[(n0 + 16) * 20 + 8 + kb2 + i];
      }
    }

    acc0 = __builtin_amdgcn_wmma_f32_16x16x32_bf16(
        false, af.v, false, bfr0.v, (short)0, acc0, false, false);
    acc1 = __builtin_amdgcn_wmma_f32_16x16x32_bf16(
        false, af.v, false, bfr1.v, (short)0, acc1, false, false);

    if (more) {
      *reinterpret_cast<uint4*>(&AsBuf[p ^ 1][ldsoff]) = ra;
      *reinterpret_cast<uint4*>(&BsBuf[p ^ 1][ldsoff]) = rb;
    }
    __syncthreads();
    p ^= 1;
  }

  // ---- epilogue (C/D layout: VGPR i -> M = 8*(lane>=16)+i, N = lane&15)
#pragma unroll
  for (int nt = 0; nt < 2; ++nt) {
    const v8f& a = nt ? acc1 : acc0;
    const int co = co_t * 64 + 32 * wn + 16 * nt + (lane & 15);

    if constexpr (MODE == 0) {
      const float inv = bnw[co] * rsqrtf(bnv[co] + 1e-5f);
      const float sh  = bnb[co] - bnm[co] * inv;
      const float bi  = bias[co];
#pragma unroll
      for (int i = 0; i < 8; ++i) {
        int m = mbase + 16 * wm + 8 * (lane >> 4) + i;
        int h = m / W, w = m % W;
        outp[(((size_t)b * C + co) * HH + h) * W + w] = (a[i] + bi) * inv + sh;
      }
    } else {
      const float bi = bias[co];
      float s = 0.f;
#pragma unroll
      for (int i = 0; i < 8; ++i) {
        int m = mbase + 16 * wm + 8 * (lane >> 4) + i;
        int h = m / W, w = m % W;
        float xv = xf32[(((size_t)b * C + co) * HH + h) * W + w];
        float yv = y1in[(((size_t)b * C + co) * (HH / 4) + (h >> 2)) * (W / 4) + (w >> 2)];
        float sg = 1.f / (1.f + __expf(-(xv + yv)));
        s += (a[i] + bi) * sg;
      }
      s += __shfl_xor(s, 16, 32);
      if (lane < 16) atomicAdd(&outp[b * C + co], s);
    }
  }
}

// ------------------------- head: t = tanh(bn(gm/4096 @ w4^T)) --------------

__global__ __launch_bounds__(256) void mk_head(
    const float* __restrict__ gm, const float* __restrict__ w4,
    const float* __restrict__ bnw, const float* __restrict__ bnb,
    const float* __restrict__ bnm, const float* __restrict__ bnv,
    float* __restrict__ t) {
  int idx = blockIdx.x * 256 + threadIdx.x;  // < 16*768
  if (idx >= 16 * 768) return;
  int j = idx % 768, b = idx / 768;
  const float* g  = gm + b * 256;
  const float* wr = w4 + j * 256;
  float s = 0.f;
#pragma unroll 4
  for (int c = 0; c < 256; ++c) s += g[c] * wr[c];
  s *= (1.f / 4096.f);
  float inv = bnw[j] * rsqrtf(bnv[j] + 1e-5f);
  t[idx] = tanhf((s - bnm[j]) * inv + bnb[j]);
}

// ------------------------- final fused elementwise -------------------------

__global__ __launch_bounds__(256) void mk_finish(
    const float* __restrict__ x, const float* __restrict__ t,
    const float* __restrict__ gamma, const float* __restrict__ beta,
    float* __restrict__ out) {
  size_t idx = (size_t)blockIdx.x * 256 + threadIdx.x;  // 16*256*64*64
  int hw = (int)(idx & 4095);
  int c  = (int)((idx >> 12) & 255);
  int b  = (int)(idx >> 20);
  float x0 = x[idx];
  int cm = (c == 0) ? 1 : c - 1;
  int cp = (c == 255) ? 254 : c + 1;
  float xm = x[(((size_t)b * 256 + cm) << 12) + hw];
  float xp = x[(((size_t)b * 256 + cp) << 12) + hw];
  const float* tb = t + b * 768;
  float r = xm * tb[c] + x0 * tb[256 + c] + xp * tb[512 + c];
  out[idx] = r * gamma[c] + x0 * beta[c];
}

// ------------------------- launcher ----------------------------------------

extern "C" void kernel_launch(void* const* d_in, const int* in_sizes, int n_in,
                              void* d_out, int out_size, void* d_ws, size_t ws_size,
                              hipStream_t stream) {
  (void)in_sizes; (void)n_in; (void)out_size; (void)ws_size;

  const float* x      = (const float*)d_in[0];
  const float* k2_w   = (const float*)d_in[1];
  const float* k2_b   = (const float*)d_in[2];
  const float* k2_bnw = (const float*)d_in[3];
  const float* k2_bnb = (const float*)d_in[4];
  const float* k2_bnm = (const float*)d_in[5];
  const float* k2_bnv = (const float*)d_in[6];
  const float* k3_w   = (const float*)d_in[7];
  const float* k3_b   = (const float*)d_in[8];
  const float* k4_w   = (const float*)d_in[9];
  const float* k4_bnw = (const float*)d_in[10];
  const float* k4_bnb = (const float*)d_in[11];
  const float* k4_bnm = (const float*)d_in[12];
  const float* k4_bnv = (const float*)d_in[13];
  const float* gamma  = (const float*)d_in[14];
  const float* beta   = (const float*)d_in[15];
  float* out = (float*)d_out;

  char* ws = (char*)d_ws;
  unsigned short* xpad  = (unsigned short*)(ws);               // 16*66*66*256*2 = 35,684,352
  unsigned short* y0pad = (unsigned short*)(ws + 35684352);    // 16*18*18*256*2 =  2,654,208
  unsigned short* wt2   = (unsigned short*)(ws + 38338560);    //                  1,179,648
  unsigned short* wt3   = (unsigned short*)(ws + 39518208);    //                  1,179,648
  float* y1 = (float*)(ws + 40697856);                         //                  4,194,304
  float* gm = (float*)(ws + 44892160);                         //                     16,384
  float* t  = (float*)(ws + 44908544);                         //                     49,152

  const int NX = 16 * 256 * 64 * 64;  // 16,777,216

  // zero padded buffers (borders), gm accumulator
  mk_zero_u32<<<(16 * 66 * 66 * 256 * 2 / 4) / 256, 256, 0, stream>>>((uint32_t*)xpad);
  mk_zero_u32<<<(16 * 18 * 18 * 256 * 2 / 4) / 256, 256, 0, stream>>>((uint32_t*)y0pad);
  mk_zero_f32<<<16, 256, 0, stream>>>(gm, 16 * 256);

  // layout transforms
  mk_x_nhwc<<<16 * 64, 256, 0, stream>>>(x, xpad);
  mk_pool_nhwc<<<16 * 16, 256, 0, stream>>>(x, y0pad);
  mk_wxform<<<(9 * 8 * 4 * 64 * 32) / 256, 256, 0, stream>>>(k2_w, wt2);
  mk_wxform<<<(9 * 8 * 4 * 64 * 32) / 256, 256, 0, stream>>>(k3_w, wt3);

  // conv2 on pooled maps (16x16): bias+BN fused -> y1 (fp32 NCHW)
  mk_conv3x3_wmma<16, 16, 0><<<dim3(4, 4, 16), 256, 0, stream>>>(
      y0pad, wt2, k2_b, k2_bnw, k2_bnb, k2_bnm, k2_bnv, nullptr, nullptr, y1);

  // conv3 on full maps (64x64): bias + sigmoid-gate + mean-reduce -> gm
  mk_conv3x3_wmma<64, 64, 1><<<dim3(4, 64, 16), 256, 0, stream>>>(
      xpad, wt3, k3_b, nullptr, nullptr, nullptr, nullptr, x, y1, gm);

  // head GEMM + bn + tanh -> t (16 x 768)
  mk_head<<<(16 * 768 + 255) / 256, 256, 0, stream>>>(
      gm, k4_w, k4_bnw, k4_bnb, k4_bnm, k4_bnv, t);

  // fused 3-tap channel mix + affine
  mk_finish<<<NX / 256, 256, 0, stream>>>(x, t, gamma, beta, out);
}
